// TitansMAC_75033078661339
// MI455X (gfx1250) — matrix operand order repaired and checked by reading, not compiled, not run on hardware
//
#include <hip/hip_runtime.h>
#include <hip/hip_bf16.h>

// ---- model constants -------------------------------------------------------
#define Dm   1024
#define Hn   16
#define HD   64
#define WIN  512
#define SEGm 256
#define PMm  16
#define NLm  4
#define Bb   2
#define SEQm 2048
#define NSm  8          // SEQ / SEG
#define CLm  528        // PM + SEG + SEG
#define BNm  (Bb*NSm)   // 16

typedef __attribute__((ext_vector_type(16))) __bf16 v16bf;
typedef __attribute__((ext_vector_type(8)))  float  v8f;

__device__ __forceinline__ unsigned short f2bf(float x) {
    unsigned u = __float_as_uint(x);
    unsigned r = (u + 0x7fffu + ((u >> 16) & 1u)) >> 16;   // round-nearest-even
    return (unsigned short)r;
}

// ---- generic batched GEMM: C = epi(alpha * A(f32->bf16) @ B(f32->bf16) + bias)
// A: [M,K] row-major (lda). B: [K,N] row-major (ldb) or transposed ([N,K] rows) if bTrans.
// batch z: off = (z/zdiv)*s?0 + (z%zdiv)*s?1 applied to A/B/C(and res).
// epi: 0 = +bias ; 1 = silu(acc+bias) ; 2 = res + acc + bias
// 256 threads = 8 waves as 4(M) x 2(N); each wave owns a 32x64 strip:
//   2 A-frags x 4 B-frags -> 8 v_wmma per k-step.
#define BLK_M 128
#define BLK_N 128
#define BLK_K 32
#define LPIT  34   // LDS row pitch (bf16 elems) -> conflict-free dword gathers

__global__ __launch_bounds__(256)
void gemm_bf16_kernel(const float* __restrict__ A, const float* __restrict__ Bm,
                      const float* __restrict__ bias, const float* __restrict__ res,
                      float* __restrict__ C,
                      int M, int N, int K, int lda, int ldb, int ldc,
                      long sA0, long sA1, long sB0, long sB1, long sC0, long sC1,
                      int zdiv, int bTrans, int epi, float alpha)
{
    __shared__ unsigned short As[BLK_M * LPIT];
    __shared__ unsigned short Bs[BLK_N * LPIT];

    const int t = threadIdx.x;
    const int z = blockIdx.z;
    const long zq = z / zdiv, zr = z % zdiv;
    A  += zq * sA0 + zr * sA1;
    Bm += zq * sB0 + zr * sB1;
    C  += zq * sC0 + zr * sC1;
    if (res) res += zq * sC0 + zr * sC1;

    const int m0 = blockIdx.x * BLK_M;
    const int n0 = blockIdx.y * BLK_N;

    const int w    = t >> 5;        // wave 0..7
    const int wm   = w & 3;         // M sub-tile (32 rows)
    const int wn   = w >> 2;        // N sub-tile (64 cols)
    const int lane = t & 31;
    const int lr   = lane & 15;
    const int half = lane >> 4;

    v8f acc[2][4] = {};

    // per-lane gather K offsets (ISA VGPR layouts)
    int kA[8], kB[8];
#pragma unroll
    for (int d = 0; d < 8; ++d) {
        int base = (d < 4) ? 2 * d : 16 + 2 * (d - 4);
        kA[d] = half * 8 + base;       // A 16x32 bf16 layout (interleaved halves)
        kB[d] = half * 8 + base;       // B 32x16 bf16 layout (interleaved halves)
    }
    const int am0 = wm * 32 + lr;

    for (int k0 = 0; k0 < K; k0 += BLK_K) {
        if (k0 + BLK_K < K) {   // prefetch next k-tiles -> global_prefetch_b8
            __builtin_prefetch(&A[(long)(m0 + (t >> 3)) * lda + k0 + BLK_K], 0, 1);
            __builtin_prefetch(bTrans ? &Bm[(long)(n0 + (t & 127)) * ldb + k0 + BLK_K]
                                      : &Bm[(long)(k0 + BLK_K + (t >> 7)) * ldb + n0], 0, 1);
        }
        __syncthreads();
        // stage A tile (128x32) as bf16
#pragma unroll
        for (int i = 0; i < 16; ++i) {
            int idx = i * 256 + t;
            int r = idx >> 5, c = idx & 31;
            int gm = m0 + r, gk = k0 + c;
            float v = (gm < M && gk < K) ? A[(long)gm * lda + gk] : 0.f;
            As[r * LPIT + c] = f2bf(v);
        }
        // stage B tile (32x128), transposed in LDS -> [n][k]
#pragma unroll
        for (int i = 0; i < 16; ++i) {
            int idx = i * 256 + t;
            int nn = idx & 127, kk = idx >> 7;
            int gn = n0 + nn, gk = k0 + kk;
            float v = 0.f;
            if (gn < N && gk < K)
                v = bTrans ? Bm[(long)gn * ldb + gk] : Bm[(long)gk * ldb + gn];
            Bs[nn * LPIT + kk] = f2bf(v);
        }
        __syncthreads();

        union bfrag { unsigned u[8]; v16bf v; };
        bfrag af[2];
#pragma unroll
        for (int mi = 0; mi < 2; ++mi)
#pragma unroll
            for (int d = 0; d < 8; ++d)
                af[mi].u[d] = *(const unsigned*)&As[(am0 + mi * 16) * LPIT + kA[d]];

#pragma unroll
        for (int nt = 0; nt < 4; ++nt) {
            bfrag bfv;
            int bn_row = wn * 64 + nt * 16 + lr;
#pragma unroll
            for (int d = 0; d < 8; ++d)
                bfv.u[d] = *(const unsigned*)&Bs[bn_row * LPIT + kB[d]];
#pragma unroll
            for (int mi = 0; mi < 2; ++mi)
                acc[mi][nt] = __builtin_amdgcn_wmma_f32_16x16x32_bf16(
                    false, af[mi].v, false, bfv.v, (short)0, acc[mi][nt], false, false);
        }
    }

    // epilogue; C fragment: lane<16 -> rows v, lane>=16 -> rows 8+v ; col = lr
#pragma unroll
    for (int nt = 0; nt < 4; ++nt) {
        int gn = n0 + wn * 64 + nt * 16 + lr;
        if (gn >= N) continue;
        float bv = bias ? bias[gn] : 0.f;
#pragma unroll
        for (int mi = 0; mi < 2; ++mi) {
#pragma unroll
            for (int v = 0; v < 8; ++v) {
                int gm = m0 + wm * 32 + mi * 16 + half * 8 + v;
                if (gm >= M) continue;
                long idx = (long)gm * ldc + gn;
                float s = acc[mi][nt][v] * alpha + bv;
                if (epi == 1)      s = s * (1.f / (1.f + __expf(-s)));
                else if (epi == 2) s += res[idx];
                C[idx] = s;
            }
        }
    }
}

// ---- LayerNorm (row of 1024); slice=1 maps compact row -> ctx[:,PM+SEG:] row
__global__ __launch_bounds__(256)
void layernorm_kernel(const float* __restrict__ in, const float* __restrict__ g,
                      const float* __restrict__ b, float* __restrict__ out, int slice)
{
    int r = blockIdx.x;
    const float* ip;
    if (slice) {
        int bn = r / SEGm, rr = r % SEGm;
        ip = in + ((long)bn * CLm + PMm + SEGm + rr) * Dm;
    } else ip = in + (long)r * Dm;
    float* op = out + (long)r * Dm;
    int t = threadIdx.x;
    float v[4], s = 0.f, ss = 0.f;
#pragma unroll
    for (int i = 0; i < 4; ++i) { v[i] = ip[t + i * 256]; s += v[i]; ss += v[i] * v[i]; }
    __shared__ float r1[256], r2[256];
    r1[t] = s; r2[t] = ss; __syncthreads();
    for (int o = 128; o > 0; o >>= 1) {
        if (t < o) { r1[t] += r1[t + o]; r2[t] += r2[t + o]; }
        __syncthreads();
    }
    float mean = r1[0] * (1.f / Dm);
    float var  = r2[0] * (1.f / Dm) - mean * mean;
    float inv  = rsqrtf(var + 1e-5f);
#pragma unroll
    for (int i = 0; i < 4; ++i) {
        int c = t + i * 256;
        op[c] = (v[i] - mean) * inv * g[c] + b[c];
    }
}

// ---- banded causal softmax over sc rows (window WIN)
__global__ __launch_bounds__(256)
void softmax_band_kernel(float* __restrict__ sc)
{
    int i = blockIdx.x;                    // query index in CL
    int z = blockIdx.y;                    // (bn*H + h)
    float* row = sc + (long)z * CLm * CLm + (long)i * CLm;
    int t = threadIdx.x;
    float v[3], mx = -INFINITY;
#pragma unroll
    for (int u = 0; u < 3; ++u) {
        int j = t + u * 256;
        float x = -INFINITY;
        if (j < CLm && j <= i && j > i - WIN) x = row[j];
        v[u] = x; mx = fmaxf(mx, x);
    }
    __shared__ float red[256];
    red[t] = mx; __syncthreads();
    for (int o = 128; o > 0; o >>= 1) { if (t < o) red[t] = fmaxf(red[t], red[t + o]); __syncthreads(); }
    mx = red[0]; __syncthreads();
    float s = 0.f;
#pragma unroll
    for (int u = 0; u < 3; ++u) {
        float e = (v[u] == -INFINITY) ? 0.f : __expf(v[u] - mx);
        v[u] = e; s += e;
    }
    red[t] = s; __syncthreads();
    for (int o = 128; o > 0; o >>= 1) { if (t < o) red[t] += red[t + o]; __syncthreads(); }
    float inv = 1.f / red[0];
#pragma unroll
    for (int u = 0; u < 3; ++u) {
        int j = t + u * 256;
        if (j < CLm) row[j] = v[u] * inv;
    }
}

// ---- assemble ctx = concat(persist, mem*(s>0), seg)
__global__ void build_ctx_kernel(const float* __restrict__ xe, const float* __restrict__ mem,
                                 const float* __restrict__ persist, float* __restrict__ ctx)
{
    long idx = (long)blockIdx.x * 256 + threadIdx.x;
    if (idx >= (long)BNm * CLm * Dm) return;
    int col = (int)(idx % Dm);
    long rem = idx / Dm;
    int row = (int)(rem % CLm);
    int bn  = (int)(rem / CLm);
    int s   = bn % NSm;
    float v;
    if (row < PMm)             v = persist[row * Dm + col];
    else if (row < PMm + SEGm) v = (s == 0) ? 0.f : mem[((long)bn * SEGm + (row - PMm)) * Dm + col];
    else                       v = xe[((long)bn * SEGm + (row - PMm - SEGm)) * Dm + col];
    ctx[idx] = v;
}

// ---------------------------------------------------------------------------
extern "C" void kernel_launch(void* const* d_in, const int* in_sizes, int n_in,
                              void* d_out, int out_size, void* d_ws, size_t ws_size,
                              hipStream_t stream) {
    (void)in_sizes; (void)n_in; (void)out_size; (void)ws_size;
    const float* x       = (const float*)d_in[0];
    const float* emb_W   = (const float*)d_in[1];
    const float* emb_b   = (const float*)d_in[2];
    const float* persist = (const float*)d_in[3];
    const float* qW      = (const float*)d_in[4];
    const float* qb      = (const float*)d_in[5];
    const float* mem_W1  = (const float*)d_in[6];
    const float* mem_b1  = (const float*)d_in[7];
    const float* mem_W2  = (const float*)d_in[8];
    const float* mem_b2  = (const float*)d_in[9];
    const float* attn_qW = (const float*)d_in[10];
    const float* attn_qb = (const float*)d_in[11];
    const float* attn_kW = (const float*)d_in[12];
    const float* attn_kb = (const float*)d_in[13];
    const float* attn_vW = (const float*)d_in[14];
    const float* attn_vb = (const float*)d_in[15];
    const float* attn_oW = (const float*)d_in[16];
    const float* attn_ob = (const float*)d_in[17];
    const float* norm_g  = (const float*)d_in[18];
    const float* norm_b  = (const float*)d_in[19];
    const float* ffn_W1  = (const float*)d_in[20];
    const float* ffn_b1  = (const float*)d_in[21];
    const float* ffn_W2  = (const float*)d_in[22];
    const float* ffn_b2  = (const float*)d_in[23];
    const float* onorm_g = (const float*)d_in[24];
    const float* onorm_b = (const float*)d_in[25];
    const float* out_W   = (const float*)d_in[26];
    const float* out_b   = (const float*)d_in[27];

    const long NTOK = (long)Bb * SEQm;     // 4096
    const long CTOK = (long)BNm * CLm;     // 8448

    float* ws   = (float*)d_ws;
    float* xe   = ws;
    float* q    = xe  + NTOK * Dm;
    float* memh = q   + NTOK * Dm;
    float* mem  = memh+ NTOK * Dm;
    float* ctx  = mem + NTOK * Dm;
    float* h    = ctx + CTOK * Dm;
    float* qh   = h   + CTOK * Dm;
    float* kh   = qh  + CTOK * Dm;
    float* vh   = kh  + CTOK * Dm;
    float* ao   = vh  + CTOK * Dm;
    float* ffnh = qh;                      // alias: qh..ao dead when ffnh is live (exact fit)
    float* sc   = ao  + CTOK * Dm;

    auto gemm = [&](const float* A, const float* Bm_, const float* bias, const float* res,
                    float* C, int M, int N, int K, int lda, int ldb, int ldc,
                    long sA0, long sA1, long sB0, long sB1, long sC0, long sC1,
                    int zcnt, int zdiv, int bT, int epi, float alpha) {
        dim3 g((M + BLK_M - 1) / BLK_M, (N + BLK_N - 1) / BLK_N, zcnt);
        gemm_bf16_kernel<<<g, 256, 0, stream>>>(A, Bm_, bias, res, C, M, N, K, lda, ldb, ldc,
                                                sA0, sA1, sB0, sB1, sC0, sC1, zdiv, bT, epi, alpha);
    };

    // embedding, q-proj, memory MLP
    gemm(x,    emb_W,  emb_b,  nullptr, xe,   (int)NTOK, Dm, Dm, Dm, Dm, Dm, 0,0,0,0,0,0, 1,1,0,0,1.f);
    gemm(xe,   qW,     qb,     nullptr, q,    (int)NTOK, Dm, Dm, Dm, Dm, Dm, 0,0,0,0,0,0, 1,1,0,0,1.f);
    gemm(q,    mem_W1, mem_b1, nullptr, memh, (int)NTOK, Dm, Dm, Dm, Dm, Dm, 0,0,0,0,0,0, 1,1,0,1,1.f);
    gemm(memh, mem_W2, mem_b2, nullptr, mem,  (int)NTOK, Dm, Dm, Dm, Dm, Dm, 0,0,0,0,0,0, 1,1,0,0,1.f);

    long tot = (long)BNm * CLm * Dm;
    build_ctx_kernel<<<(unsigned)((tot + 255) / 256), 256, 0, stream>>>(xe, mem, persist, ctx);

    const float scale = 0.125f;  // 64^-0.5
    const long  sBH   = (long)CLm * Dm;     // per-(bn) stride into qh/kh/vh
    const long  sSC   = (long)CLm * CLm;    // per-head stride into scores

    for (int j = 0; j < NLm; ++j) {
        layernorm_kernel<<<(unsigned)CTOK, 256, 0, stream>>>(ctx, norm_g + j * Dm, norm_b + j * Dm, h, 0);
        gemm(h, attn_qW + (long)j * Dm * Dm, attn_qb + j * Dm, nullptr, qh, (int)CTOK, Dm, Dm, Dm, Dm, Dm, 0,0,0,0,0,0, 1,1,0,0,1.f);
        gemm(h, attn_kW + (long)j * Dm * Dm, attn_kb + j * Dm, nullptr, kh, (int)CTOK, Dm, Dm, Dm, Dm, Dm, 0,0,0,0,0,0, 1,1,0,0,1.f);
        gemm(h, attn_vW + (long)j * Dm * Dm, attn_vb + j * Dm, nullptr, vh, (int)CTOK, Dm, Dm, Dm, Dm, Dm, 0,0,0,0,0,0, 1,1,0,0,1.f);
        // scores = scale * Q @ K^T  (batched over 16 bn x 16 heads)
        gemm(qh, kh, nullptr, nullptr, sc, CLm, CLm, HD, Dm, Dm, CLm,
             sBH, HD, sBH, HD, (long)Hn * sSC, sSC, BNm * Hn, Hn, 1, 0, scale);
        { dim3 sg(CLm, BNm * Hn); softmax_band_kernel<<<sg, 256, 0, stream>>>(sc); }
        // ao = P @ V (batched)
        gemm(sc, vh, nullptr, nullptr, ao, CLm, HD, CLm, CLm, Dm, Dm,
             (long)Hn * sSC, sSC, sBH, HD, sBH, HD, BNm * Hn, Hn, 0, 0, 1.f);
        // ctx += ao @ oW + ob
        gemm(ao, attn_oW + (long)j * Dm * Dm, attn_ob + j * Dm, ctx, ctx, (int)CTOK, Dm, Dm, Dm, Dm, Dm, 0,0,0,0,0,0, 1,1,0,2,1.f);
        // FFN
        layernorm_kernel<<<(unsigned)CTOK, 256, 0, stream>>>(ctx, norm_g + j * Dm, norm_b + j * Dm, h, 0);
        gemm(h, ffn_W1 + (long)j * Dm * 4 * Dm, ffn_b1 + (long)j * 4 * Dm, nullptr, ffnh,
             (int)CTOK, 4 * Dm, Dm, Dm, 4 * Dm, 4 * Dm, 0,0,0,0,0,0, 1,1,0,1,1.f);
        gemm(ffnh, ffn_W2 + (long)j * 4 * Dm * Dm, ffn_b2 + j * Dm, ctx, ctx,
             (int)CTOK, Dm, 4 * Dm, 4 * Dm, Dm, Dm, 0,0,0,0,0,0, 1,1,0,2,1.f);
    }

    // out = LN(ctx[:, PM+SEG:]) @ out_W + out_b   (compact row order == output row order)
    layernorm_kernel<<<(unsigned)NTOK, 256, 0, stream>>>(ctx, onorm_g, onorm_b, h, 1);
    gemm(h, out_W, out_b, nullptr, (float*)d_out, (int)NTOK, Dm, Dm, Dm, Dm, Dm, 0,0,0,0,0,0, 1,1,0,0,1.f);

    // second tuple output: mem_W2 passthrough
    hipMemcpyAsync((float*)d_out + NTOK * Dm, d_in[8], (size_t)Dm * Dm * sizeof(float),
                   hipMemcpyDeviceToDevice, stream);
}
